// RandlaNet_65558380806321
// MI455X (gfx1250) — compile-verified, matrix operand order checked by reading, not run on hardware
//
#include <hip/hip_runtime.h>
#include <hip/hip_bf16.h>

// ---------------------------------------------------------------------------
// RandLA-Net forward for MI455X (gfx1250).
// All linear layers run through a bf16 WMMA GEMM (v_wmma_f32_16x16x32_bf16,
// fp32 accumulate). Weight tiles are staged into LDS with the CDNA5 async
// global->LDS copy path (ASYNCcnt + s_wait_asynccnt). Gathers / BN /
// softmax-pool are bandwidth-bound elementwise kernels. The net is
// memory-bound (~0.15 TFLOP vs GBs of gather traffic at 23.3 TB/s), so the
// GEMM targets coalesced loads + matrix-op density rather than deep software
// pipelining.
// ---------------------------------------------------------------------------

#define LEAK 0.2f
#define BN_EPS 1e-5f

typedef __attribute__((ext_vector_type(16))) __bf16          v16bf;
typedef __attribute__((ext_vector_type(16))) unsigned short  v16us;
typedef __attribute__((ext_vector_type(8)))  float           v8f;

__device__ __forceinline__ float dleaky(float x) { return x > 0.f ? x : LEAK * x; }

// float -> bf16 (round to nearest even), as raw u16
__device__ __forceinline__ unsigned short f2bf(float f) {
  union { float f; unsigned u; } c; c.f = f;
  unsigned u = c.u;
  return (unsigned short)((u + 0x7FFFu + ((u >> 16) & 1u)) >> 16);
}

// ---------------------------------------------------------------------------
// WMMA GEMM:  C[M,N] = act(A[M,K] @ W[K,N] + bias)
// One wave computes a 16x32 C tile (two 16x16 WMMA accumulators); 8 waves per
// block stacked along M => block covers 128(M) x 32(N). K tiled by 32.
//
// W tile (32K x 32N fp32) staged in LDS. On the block-uniform fast path
// (full tile in bounds) the stage uses global_load_async_to_lds_b32
// (ASYNCcnt, drained by s_wait_asynccnt before the workgroup barrier);
// edge tiles fall back to guarded ds stores.
//
// A fragment (16-bit A 16x32 layout, ISA 7.12.2): lane L holds row M = L%16;
// with half = L/16, vector elements j=0..7  -> K = kk + half*8 + j,
//                              j=8..15 -> K = kk + 16 + half*8 + (j-8)
// i.e. two contiguous 8-wide runs along K => coalesced loads on fast path.
// B fragment mirrored (lane holds column N = L%16 of the subtile, same K
// packing). C/D layout: VGPR r: lanes0-15 -> M=r, lanes16-31 -> M=8+r.
// ---------------------------------------------------------------------------
__global__ void __launch_bounds__(256)
k_gemm(const float* __restrict__ A, const float* __restrict__ W,
       const float* __restrict__ bias, float* __restrict__ C,
       int M, int K, int N, int act)
{
#if defined(__HIP_DEVICE_COMPILE__)
  __shared__ float Wl[32 * 32];   // [k][n] fp32 weight tile

  const int tid  = threadIdx.x;
  const int wave = tid >> 5;
  const int lane = tid & 31;
  const int half = lane >> 4;
  const int l16  = lane & 15;
  const int n0   = blockIdx.y * 32;
  const int m0   = (blockIdx.x * 8 + wave) * 16;
  const int row  = m0 + l16;

  v8f acc0 = {};
  v8f acc1 = {};

  for (int kk = 0; kk < K; kk += 32) {
    const bool fullTile = (kk + 32 <= K) && (n0 + 32 <= N);  // block-uniform
    if (fullTile) {
      // Async global->LDS stage: 1024 fp32 elements, 4 per thread.
#pragma unroll
      for (int i = 0; i < 4; ++i) {
        int e = tid + i * 256;
        int k = e >> 5, n = e & 31;
        const float* gp = W + (size_t)(kk + k) * N + (n0 + n);
        unsigned lds = (unsigned)(size_t)(&Wl[e]);
        asm volatile("global_load_async_to_lds_b32 %0, %1, off"
                     :: "v"(lds), "v"((unsigned long long)(size_t)gp)
                     : "memory");
      }
      asm volatile("s_wait_asynccnt 0x0" ::: "memory");
    } else {
      for (int e = tid; e < 1024; e += 256) {
        int k = e >> 5, n = e & 31;
        float w = 0.f;
        if (kk + k < K && n0 + n < N) w = W[(size_t)(kk + k) * N + (n0 + n)];
        Wl[e] = w;
      }
    }
    __syncthreads();

    union { v16us u; v16bf b; } af, b0, b1;
    const bool fullK = (kk + 32 <= K);
    if (row < M && fullK) {
      // contiguous 8-wide runs -> coalesced; prefetch next K tile
      const float* ap = A + (size_t)row * K + kk + half * 8;
      __builtin_prefetch(ap + 32, 0, 1);
#pragma unroll
      for (int j = 0; j < 8; ++j) {
        af.u[j]     = f2bf(ap[j]);
        af.u[j + 8] = f2bf(ap[j + 16]);
      }
    } else {
#pragma unroll
      for (int j = 0; j < 8; ++j) {
        int ka = kk + half * 8 + j;
        int kb = ka + 16;
        float a0 = (row < M && ka < K) ? A[(size_t)row * K + ka] : 0.f;
        float a1 = (row < M && kb < K) ? A[(size_t)row * K + kb] : 0.f;
        af.u[j]     = f2bf(a0);
        af.u[j + 8] = f2bf(a1);
      }
    }

#pragma unroll
    for (int j = 0; j < 8; ++j) {
      int kA = (half * 8 + j) * 32;        // first K run, LDS row offset
      int kB = (16 + half * 8 + j) * 32;   // second K run
      b0.u[j]     = f2bf(Wl[kA + l16]);
      b0.u[j + 8] = f2bf(Wl[kB + l16]);
      b1.u[j]     = f2bf(Wl[kA + 16 + l16]);
      b1.u[j + 8] = f2bf(Wl[kB + 16 + l16]);
    }

    acc0 = __builtin_amdgcn_wmma_f32_16x16x32_bf16(
        false, af.b, false, b0.b, (short)0, acc0, false, false);
    acc1 = __builtin_amdgcn_wmma_f32_16x16x32_bf16(
        false, af.b, false, b1.b, (short)0, acc1, false, false);

    __syncthreads();
  }

  // store 16x32 tile
#pragma unroll
  for (int t = 0; t < 2; ++t) {
    const int n = n0 + t * 16 + l16;
    if (n < N) {
      float bv = bias ? bias[n] : 0.f;
#pragma unroll
      for (int r = 0; r < 8; ++r) {
        int m = m0 + half * 8 + r;
        if (m < M) {
          float v = (t == 0 ? acc0[r] : acc1[r]) + bv;
          if (act == 1) v = dleaky(v);
          C[(size_t)m * N + n] = v;
        }
      }
    }
  }
#else
  (void)A; (void)W; (void)bias; (void)C; (void)M; (void)K; (void)N; (void)act;
#endif
}

// ---------------------------------------------------------------------------
// Elementwise / gather / reduction kernels (grid-stride).
// ---------------------------------------------------------------------------
__global__ void k_zero(float* __restrict__ p, long long n) {
  for (long long i = blockIdx.x * (long long)blockDim.x + threadIdx.x; i < n;
       i += (long long)gridDim.x * blockDim.x) p[i] = 0.f;
}

// out[b,m,kq, dstOff:dstOff+C] = feat[b, idx[b,m,kq], 0:C]; out row stride dstStride
__global__ void k_gather(const float* __restrict__ feat, const int* __restrict__ idx,
                         float* __restrict__ out, long long total,
                         int Nin, int M, int Kk, int C, int dstStride, int dstOff)
{
  for (long long i = blockIdx.x * (long long)blockDim.x + threadIdx.x; i < total;
       i += (long long)gridDim.x * blockDim.x) {
    int c = (int)(i % C); long long t = i / C;
    int kq = (int)(t % Kk); long long t2 = t / Kk;
    int m = (int)(t2 % M); int b = (int)(t2 / M);
    int src = idx[((long long)b * M + m) * Kk + kq];
    out[t * dstStride + dstOff + c] = feat[((long long)b * Nin + src) * C + c];
  }
}

// dst[r, dstOff:dstOff+C] = src[r, 0:C]
__global__ void k_copys(const float* __restrict__ src, float* __restrict__ dst,
                        long long R, int C, int dstStride, int dstOff)
{
  long long total = R * (long long)C;
  for (long long i = blockIdx.x * (long long)blockDim.x + threadIdx.x; i < total;
       i += (long long)gridDim.x * blockDim.x) {
    long long r = i / C; int c = (int)(i % C);
    dst[r * dstStride + dstOff + c] = src[i];
  }
}

// out[b,m,c] = max_k feat[b, idx[b,m,k], c]
__global__ void k_gmax(const float* __restrict__ feat, const int* __restrict__ idx,
                       float* __restrict__ out, long long total,
                       int Nin, int M, int Kk, int C)
{
  for (long long i = blockIdx.x * (long long)blockDim.x + threadIdx.x; i < total;
       i += (long long)gridDim.x * blockDim.x) {
    int c = (int)(i % C); long long t = i / C;
    int m = (int)(t % M); int b = (int)(t / M);
    const int* ip = idx + ((long long)b * M + m) * Kk;
    float mx = -3.4e38f;
    for (int k = 0; k < Kk; ++k)
      mx = fmaxf(mx, feat[((long long)b * Nin + ip[k]) * C + c]);
    out[i] = mx;
  }
}

// pe[b,n,k,c] = xyz[b,n,c] - xyz[b, idx[b,n,k], c]   (C == 3)
__global__ void k_pe(const float* __restrict__ xyz, const int* __restrict__ idx,
                     float* __restrict__ pe, long long total, int N, int Kk)
{
  for (long long i = blockIdx.x * (long long)blockDim.x + threadIdx.x; i < total;
       i += (long long)gridDim.x * blockDim.x) {
    int c = (int)(i % 3); long long t = i / 3;
    int kq = (int)(t % Kk); long long t2 = t / Kk;
    int n = (int)(t2 % N); int b = (int)(t2 / N);
    int s = idx[((long long)b * N + n) * Kk + kq];
    pe[i] = xyz[((long long)b * N + n) * 3 + c] - xyz[((long long)b * N + s) * 3 + c];
  }
}

// BN pass 1: per-channel sum & sumsq (LDS bins -> global atomics). C <= 512.
__global__ void k_bnstats(const float* __restrict__ x, float* __restrict__ sums,
                          long long R, int C)
{
  __shared__ float s[1024];
  for (int i = threadIdx.x; i < 2 * C; i += blockDim.x) s[i] = 0.f;
  __syncthreads();
  long long total = R * (long long)C;
  for (long long i = blockIdx.x * (long long)blockDim.x + threadIdx.x; i < total;
       i += (long long)gridDim.x * blockDim.x) {
    float v = x[i]; int c = (int)(i % C);
    atomicAdd(&s[c], v);
    atomicAdd(&s[C + c], v * v);
  }
  __syncthreads();
  for (int i = threadIdx.x; i < 2 * C; i += blockDim.x) atomicAdd(&sums[i], s[i]);
}

// BN pass 2: scale = g * rsqrt(var+eps), shift = b - mean*scale
__global__ void k_bnfin(const float* __restrict__ sums, const float* __restrict__ g,
                        const float* __restrict__ b, float* __restrict__ ss,
                        int C, float invR)
{
  int c = blockIdx.x * blockDim.x + threadIdx.x;
  if (c >= C) return;
  float mean = sums[c] * invR;
  float var  = sums[C + c] * invR - mean * mean;
  float sc   = g[c] * rsqrtf(var + BN_EPS);
  ss[c] = sc;
  ss[C + c] = b[c] - mean * sc;
}

// BN pass 3: y = x*scale + shift, optional leaky
__global__ void k_bnapply(const float* __restrict__ x, const float* __restrict__ ss,
                          float* __restrict__ y, long long total, int C, int act)
{
  for (long long i = blockIdx.x * (long long)blockDim.x + threadIdx.x; i < total;
       i += (long long)gridDim.x * blockDim.x) {
    int c = (int)(i % C);
    float v = x[i] * ss[c] + ss[C + c];
    if (act) v = dleaky(v);
    y[i] = v;
  }
}

__global__ void k_addleaky(const float* __restrict__ a, const float* __restrict__ b,
                           float* __restrict__ y, long long n)
{
  for (long long i = blockIdx.x * (long long)blockDim.x + threadIdx.x; i < n;
       i += (long long)gridDim.x * blockDim.x) y[i] = dleaky(a[i] + b[i]);
}

// attention pool core: softmax over K of logits, weighted sum of x.
// x, logits: (R, K, C); out: (R, C)
__global__ void k_smaxpool(const float* __restrict__ x, const float* __restrict__ lg,
                           float* __restrict__ out, long long R, int Kk, int C)
{
  long long total = R * (long long)C;
  for (long long i = blockIdx.x * (long long)blockDim.x + threadIdx.x; i < total;
       i += (long long)gridDim.x * blockDim.x) {
    long long r = i / C; int c = (int)(i % C);
    const float* lp = lg + (r * Kk) * (long long)C + c;
    const float* xp = x  + (r * Kk) * (long long)C + c;
    float mx = -3.4e38f;
    for (int k = 0; k < Kk; ++k) mx = fmaxf(mx, lp[(long long)k * C]);
    float se = 0.f, ws = 0.f;
    for (int k = 0; k < Kk; ++k) {
      float e = expf(lp[(long long)k * C] - mx);
      se += e;
      ws += e * xp[(long long)k * C];
    }
    out[i] = ws / se;
  }
}

// ---------------------------------------------------------------------------
// Host-side orchestration
// ---------------------------------------------------------------------------
static inline long long cdivll(long long a, long long b) { return (a + b - 1) / b; }
static inline unsigned eblocks(long long n) {
  long long b = cdivll(n, 256);
  if (b > 32768) b = 32768;
  if (b < 1) b = 1;
  return (unsigned)b;
}

static void gemm(hipStream_t s, const float* A, const float* W, const float* b,
                 float* C, long long M, int K, int N, int act)
{
  dim3 grid((unsigned)cdivll(M, 128), (unsigned)((N + 31) / 32));
  k_gemm<<<grid, 256, 0, s>>>(A, W, b, C, (int)M, K, N, act);
}

static void run_bn(hipStream_t s, const float* x, const float* g, const float* b,
                   float* y, long long R, int C, int act, float* scratch)
{
  float* sums = scratch;          // 2C
  float* ss   = scratch + 2 * C;  // 2C
  k_zero<<<1, 256, 0, s>>>(sums, 2 * (long long)C);
  long long total = R * (long long)C;
  unsigned sb = eblocks(total); if (sb > 1024) sb = 1024;
  k_bnstats<<<sb, 256, 0, s>>>(x, sums, R, C);
  k_bnfin<<<(unsigned)((C + 127) / 128), 128, 0, s>>>(sums, g, b, ss, C, 1.0f / (float)R);
  k_bnapply<<<eblocks(total), 256, 0, s>>>(x, ss, y, total, C, act);
}

extern "C" void kernel_launch(void* const* d_in, const int* in_sizes, int n_in,
                              void* d_out, int out_size, void* d_ws, size_t ws_size,
                              hipStream_t stream)
{
  (void)in_sizes; (void)n_in; (void)out_size; (void)ws_size;

  static const int NS[5]   = {40960, 10240, 2560, 640, 160};
  static const int DOUT[4] = {16, 64, 128, 256};
  const int Bsz = 2, KN = 16;

  auto F = [&](int& i) -> const float* { return (const float*)d_in[i++]; };
  auto I = [&](int& i) -> const int*   { return (const int*)d_in[i++]; };

  // ---- parameter leaves in pytree order (dict keys sorted, lists in order)
  int pi = 0;
  const float *bridge_W = F(pi), *bridge_b = F(pi);
  const float *c2f_W = F(pi), *c2f_b = F(pi);

  struct DecP { const float *bn1_b,*bn1_g,*fc1_W,*fc1_b,*sk_W,*sk_b,*sbn_b,*sbn_g; } dp[4];
  for (int j = 0; j < 4; ++j) {
    dp[j].bn1_b = F(pi); dp[j].bn1_g = F(pi);
    dp[j].fc1_W = F(pi); dp[j].fc1_b = F(pi);
    dp[j].sk_W  = F(pi); dp[j].sk_b  = F(pi);
    dp[j].sbn_b = F(pi); dp[j].sbn_g = F(pi);
  }

  struct AttP { const float *aW,*ab,*bnb,*bng,*oW,*ob; };
  struct EncP {
    const float *bn1_b,*bn1_g,*bn2_b,*bn2_g;
    AttP a1, a2;
    const float *bf1_W,*bf1_b,*bf2_W,*bf2_b;
    const float *fc1_W,*fc1_b,*fc2_W,*fc2_b,*sc_W,*sc_b,*scbn_b,*scbn_g;
  } ep[4];
  for (int l = 0; l < 4; ++l) {
    ep[l].bn1_b = F(pi); ep[l].bn1_g = F(pi);
    ep[l].bn2_b = F(pi); ep[l].bn2_g = F(pi);
    // block: att1 (attn{W,b}, bn{b,g}, out{W,b}), att2, fc1, fc2
    ep[l].a1.aW = F(pi); ep[l].a1.ab = F(pi);
    ep[l].a1.bnb = F(pi); ep[l].a1.bng = F(pi);
    ep[l].a1.oW = F(pi); ep[l].a1.ob = F(pi);
    ep[l].a2.aW = F(pi); ep[l].a2.ab = F(pi);
    ep[l].a2.bnb = F(pi); ep[l].a2.bng = F(pi);
    ep[l].a2.oW = F(pi); ep[l].a2.ob = F(pi);
    ep[l].bf1_W = F(pi); ep[l].bf1_b = F(pi);
    ep[l].bf2_W = F(pi); ep[l].bf2_b = F(pi);
    ep[l].fc1_W = F(pi); ep[l].fc1_b = F(pi);
    ep[l].fc2_W = F(pi); ep[l].fc2_b = F(pi);
    ep[l].sc_W  = F(pi); ep[l].sc_b  = F(pi);
    ep[l].scbn_b = F(pi); ep[l].scbn_g = F(pi);
  }
  const float *h0W = F(pi), *h0b = F(pi);
  const float *h1W = F(pi), *h1b = F(pi);
  const float *h2W = F(pi), *h2b = F(pi);

  const float* features = F(pi);
  const float* xyz[4]; const int *nei[4], *sub[4], *itp[4];
  for (int l = 0; l < 4; ++l) {
    xyz[l] = F(pi); nei[l] = I(pi); sub[l] = I(pi); itp[l] = I(pi);
  }

  // ---- workspace: persistent bump from front; temp region (reset per stage)
  char* wsb = (char*)d_ws;
  size_t poff = 0;
  const size_t TEMP_BASE = (size_t)256 << 20;  // persistent fits well under 256MB
  size_t toff = TEMP_BASE;
  auto allocP = [&](long long nf) -> float* {
    float* p = (float*)(wsb + poff);
    poff += ((size_t)nf * 4 + 255) & ~(size_t)255;
    return p;
  };
  auto allocT = [&](long long nf) -> float* {
    float* p = (float*)(wsb + toff);
    toff += ((size_t)nf * 4 + 255) & ~(size_t)255;
    return p;
  };
  auto treset = [&]() { toff = TEMP_BASE; };

  float* bnscr = allocP(4 * 512);  // BN scratch, C<=512

  hipStream_t s = stream;

  // att_pool: xcat (R,K,C2) -> out (R,Cout)
  auto att_pool = [&](const float* xcat, long long R, int Cout, int C2,
                      const AttP& ap, float* out,
                      float* t_logits, float* t_pool, float* t_y) {
    long long RK = R * KN;
    gemm(s, xcat, ap.aW, ap.ab, t_logits, RK, C2, C2, 0);
    k_smaxpool<<<eblocks(R * (long long)C2), 256, 0, s>>>(xcat, t_logits, t_pool, R, KN, C2);
    gemm(s, t_pool, ap.oW, ap.ob, t_y, R, C2, Cout, 0);
    run_bn(s, t_y, ap.bng, ap.bnb, out, R, Cout, 1, bnscr);
  };

  // dilated residual block for level l: x (B,N,fin) -> out (B,N,2*fout)
  auto drb = [&](int l, const float* x, int fin, float* out) {
    const int fo = DOUT[l], C2 = 2 * fo, N = NS[l];
    const long long R = (long long)Bsz * N, RK = R * KN;
    treset();

    float* f1 = allocT(R * fo);
    gemm(s, x, ep[l].fc1_W, ep[l].fc1_b, f1, R, fin, fo, 0);
    run_bn(s, f1, ep[l].bn1_g, ep[l].bn1_b, f1, R, fo, 1, bnscr);

    // building block
    float* pe = allocT(RK * 3);
    k_pe<<<eblocks(RK * 3), 256, 0, s>>>(xyz[l], nei[l], pe, RK * 3, N, KN);
    float* fxyz = allocT(RK * fo);
    gemm(s, pe, ep[l].bf1_W, ep[l].bf1_b, fxyz, RK, 3, fo, 1);

    float* cat    = allocT(RK * C2);
    float* logits = allocT(RK * C2);
    float* pooled = allocT(R * C2);
    float* yb     = allocT(R * fo);
    float* agg1   = allocT(R * fo);

    k_gather<<<eblocks(RK * fo), 256, 0, s>>>(f1, nei[l], cat, RK * (long long)fo,
                                              N, N, KN, fo, C2, 0);
    k_copys<<<eblocks(RK * fo), 256, 0, s>>>(fxyz, cat, RK, fo, C2, fo);
    att_pool(cat, R, fo, C2, ep[l].a1, agg1, logits, pooled, yb);

    float* fxyz2 = allocT(RK * fo);
    gemm(s, fxyz, ep[l].bf2_W, ep[l].bf2_b, fxyz2, RK, fo, fo, 1);

    k_gather<<<eblocks(RK * fo), 256, 0, s>>>(agg1, nei[l], cat, RK * (long long)fo,
                                              N, N, KN, fo, C2, 0);
    k_copys<<<eblocks(RK * fo), 256, 0, s>>>(fxyz2, cat, RK, fo, C2, fo);
    float* f2 = allocT(R * fo);
    att_pool(cat, R, fo, C2, ep[l].a2, f2, logits, pooled, yb);

    // fc2+bn2, shortcut+bn, leaky(sum)
    float* f3 = allocT(R * C2);
    float* sc = allocT(R * C2);
    gemm(s, f2, ep[l].fc2_W, ep[l].fc2_b, f3, R, fo, C2, 0);
    run_bn(s, f3, ep[l].bn2_g, ep[l].bn2_b, f3, R, C2, 0, bnscr);
    gemm(s, x, ep[l].sc_W, ep[l].sc_b, sc, R, fin, C2, 0);
    run_bn(s, sc, ep[l].scbn_g, ep[l].scbn_b, sc, R, C2, 0, bnscr);
    k_addleaky<<<eblocks(R * C2), 256, 0, s>>>(f3, sc, out, R * (long long)C2);
  };

  // ---- encoder
  const long long R0 = (long long)Bsz * NS[0];
  float* x0 = allocP(R0 * 8);
  gemm(s, features, c2f_W, c2f_b, x0, R0, 3, 8, 1);

  float* skips[4];
  const float* xin = x0;
  int fin = 8;
  for (int l = 0; l < 4; ++l) {
    skips[l] = allocP((long long)Bsz * NS[l] * 2 * DOUT[l]);
    drb(l, xin, fin, skips[l]);
    const int C2 = 2 * DOUT[l];
    float* xs = allocP((long long)Bsz * NS[l + 1] * C2);
    long long tot = (long long)Bsz * NS[l + 1] * C2;
    k_gmax<<<eblocks(tot), 256, 0, s>>>(skips[l], sub[l], xs, tot,
                                        NS[l], NS[l + 1], KN, C2);
    xin = xs; fin = C2;
  }

  // ---- bridge
  float* xb = allocP((long long)Bsz * NS[4] * 512);
  gemm(s, xin, bridge_W, bridge_b, xb, (long long)Bsz * NS[4], 512, 512, 1);

  // ---- decoder
  const float* xd = xb;
  int Cin = 512;
  for (int j = 0; j < 4; ++j) {
    const int lvl = 3 - j, fo = DOUT[lvl], C2 = 2 * fo;
    const long long R = (long long)Bsz * NS[lvl];
    const int comb = Cin + C2;
    treset();

    float* xc = allocT(R * comb);
    k_gather<<<eblocks(R * Cin), 256, 0, s>>>(xd, itp[lvl], xc, R * (long long)Cin,
                                              NS[lvl + 1], NS[lvl], 1, Cin, comb, 0);
    k_copys<<<eblocks(R * C2), 256, 0, s>>>(skips[lvl], xc, R, C2, comb, Cin);

    float* t1 = allocT(R * C2);
    float* t2 = allocT(R * C2);
    gemm(s, xc, dp[j].fc1_W, dp[j].fc1_b, t1, R, comb, C2, 0);
    run_bn(s, t1, dp[j].bn1_g, dp[j].bn1_b, t1, R, C2, 0, bnscr);
    gemm(s, xc, dp[j].sk_W, dp[j].sk_b, t2, R, comb, C2, 0);
    run_bn(s, t2, dp[j].sbn_g, dp[j].sbn_b, t2, R, C2, 0, bnscr);

    float* xn = allocP(R * C2);
    k_addleaky<<<eblocks(R * C2), 256, 0, s>>>(t1, t2, xn, R * (long long)C2);
    xd = xn; Cin = C2;
  }

  // ---- head: 32 -> 64 -> 32 -> 13
  treset();
  float* hh1 = allocT(R0 * 64);
  float* hh2 = allocT(R0 * 32);
  gemm(s, xd, h0W, h0b, hh1, R0, 32, 64, 1);
  gemm(s, hh1, h1W, h1b, hh2, R0, 64, 32, 1);
  gemm(s, hh2, h2W, h2b, (float*)d_out, R0, 32, 13, 0);
}
